// importance_classifier_84859963835036
// MI455X (gfx1250) — matrix operand order repaired and checked by reading, not compiled, not run on hardware
//
#include <hip/hip_runtime.h>
#include <hip/hip_bf16.h>
#include <math.h>

// ---------------- types for WMMA fragments ----------------
typedef __attribute__((ext_vector_type(16))) __bf16 v16bf;
typedef __attribute__((ext_vector_type(8)))  __bf16 bf16x8;
typedef __attribute__((ext_vector_type(8)))  float  v8f;

// Problem constants
#define BATCH 2
#define CIN   64
#define DDIM  8
#define NSPAT 2048            // D*H*W
#define MROWS 4096            // BATCH*NSPAT
#define C1    32              // k/q/v channels
#define K1    1728            // 64*27
#define N1    96              // 3*32 fused output channels
#define C2    64
#define K2    864             // 32*27

// ---------------- packing kernels ----------------
__global__ void pack_w1(const float* __restrict__ wk, const float* __restrict__ wq,
                        const float* __restrict__ wv, __bf16* __restrict__ Wt) {
    int idx = blockIdx.x * 256 + threadIdx.x;
    if (idx >= N1 * K1) return;
    int c = idx / K1, k = idx % K1;
    const float* src = (c < 32) ? wk : (c < 64 ? wq : wv);
    int co = c & 31;
    Wt[idx] = (__bf16)src[co * K1 + k];   // [O][I][27] flat == co*K1 + k
}

__global__ void pack_w2(const float* __restrict__ wa, __bf16* __restrict__ Wt) {
    int idx = blockIdx.x * 256 + threadIdx.x;
    if (idx >= C2 * K2) return;
    Wt[idx] = (__bf16)wa[idx];            // [64][32*27] already [col][k]
}

// ---------------- im2col (fp32 -> bf16) ----------------
__global__ void im2col1(const float* __restrict__ x, __bf16* __restrict__ A) {
    int idx = blockIdx.x * 256 + threadIdx.x;
    if (idx >= MROWS * K1) return;
    int p = idx / K1, k = idx % K1;
    int b = p >> 11, n = p & 2047;
    int d = n >> 8, h = (n >> 4) & 15, w = n & 15;
    int ci = k / 27, r = k % 27;
    int kd = r / 9, kh = (r / 3) % 3, kw = r % 3;
    int dd = d + kd - 1, hh = h + kh - 1, ww = w + kw - 1;
    float v = 0.f;
    if ((unsigned)dd < 8u && (unsigned)hh < 16u && (unsigned)ww < 16u)
        v = x[((b * CIN + ci) * DDIM + dd) * 256 + hh * 16 + ww];
    A[idx] = (__bf16)v;
}

// y = attention output [C,N,B] buffer reinterpreted as flat [B][32][2048]
__global__ void im2col2(const float* __restrict__ y, __bf16* __restrict__ A) {
    int idx = blockIdx.x * 256 + threadIdx.x;
    if (idx >= MROWS * K2) return;
    int p = idx / K2, k = idx % K2;
    int b = p >> 11, n = p & 2047;
    int d = n >> 8, h = (n >> 4) & 15, w = n & 15;
    int ci = k / 27, r = k % 27;
    int kd = r / 9, kh = (r / 3) % 3, kw = r % 3;
    int dd = d + kd - 1, hh = h + kh - 1, ww = w + kw - 1;
    float v = 0.f;
    if ((unsigned)dd < 8u && (unsigned)hh < 16u && (unsigned)ww < 16u)
        v = y[(b * C1 + ci) * NSPAT + dd * 256 + hh * 16 + ww];
    A[idx] = (__bf16)v;
}

// ---- fragment loaders (16-bit WMMA register layouts, wave32) ----
__device__ __forceinline__ v16bf load_afrag(const __bf16* p, int kb, int half) {
    // A 16x32: lanes0-15 elems0-7 -> K0..7, elems8-15 -> K16..23; lanes16-31 +8
    union { v16bf v; bf16x8 h[2]; } u;
    u.h[0] = *(const bf16x8*)(p + kb + half * 8);
    u.h[1] = *(const bf16x8*)(p + kb + 16 + half * 8);
    return u.v;
}
__device__ __forceinline__ v16bf load_bfrag(const __bf16* p, int kb, int half) {
    // B 32x16: lanes0-15 elems0-7 -> K0..7, elems8-15 -> K16..23; lanes16-31 +8
    union { v16bf v; bf16x8 h[2]; } u;
    u.h[0] = *(const bf16x8*)(p + kb + half * 8);
    u.h[1] = *(const bf16x8*)(p + kb + 16 + half * 8);
    return u.v;
}

// ---------------- WMMA GEMM 1: [4096 x 1728] x [1728 x 96] -> q/k/v ----------------
// 2x2 register tiling: each wave computes a 32x32 output tile (4 WMMAs per 8 b128 loads).
__global__ void gemm1_qkv(const __bf16* __restrict__ A, const __bf16* __restrict__ Bt,
                          const float* __restrict__ bk, const float* __restrict__ bq,
                          const float* __restrict__ bv,
                          float* __restrict__ Kb, float* __restrict__ Qb, float* __restrict__ Vb) {
    int wave = threadIdx.x >> 5, lane = threadIdx.x & 31;
    int half = lane >> 4,        lcol = lane & 15;
    int tileM = (blockIdx.x * 8 + wave) * 32;
    int tileN = blockIdx.y * 32;

    const __bf16* pA0 = A  + (size_t)(tileM + lcol) * K1;
    const __bf16* pA1 = A  + (size_t)(tileM + 16 + lcol) * K1;
    const __bf16* pB0 = Bt + (size_t)(tileN + lcol) * K1;
    const __bf16* pB1 = Bt + (size_t)(tileN + 16 + lcol) * K1;

    v8f acc[2][2] = {{{}, {}}, {{}, {}}};
    for (int kb = 0; kb < K1; kb += 32) {
        v16bf a0 = load_afrag(pA0, kb, half);
        v16bf a1 = load_afrag(pA1, kb, half);
        v16bf b0 = load_bfrag(pB0, kb, half);
        v16bf b1 = load_bfrag(pB1, kb, half);
        __builtin_prefetch(pA0 + kb + 256, 0, 0);
        __builtin_prefetch(pA1 + kb + 256, 0, 0);
        acc[0][0] = __builtin_amdgcn_wmma_f32_16x16x32_bf16(false, a0, false, b0, (short)0, acc[0][0], false, false);
        acc[0][1] = __builtin_amdgcn_wmma_f32_16x16x32_bf16(false, a0, false, b1, (short)0, acc[0][1], false, false);
        acc[1][0] = __builtin_amdgcn_wmma_f32_16x16x32_bf16(false, a1, false, b0, (short)0, acc[1][0], false, false);
        acc[1][1] = __builtin_amdgcn_wmma_f32_16x16x32_bf16(false, a1, false, b1, (short)0, acc[1][1], false, false);
    }

#pragma unroll
    for (int ni = 0; ni < 2; ++ni) {
        int cOut = tileN + ni * 16 + lcol;
        float bias = (cOut < 32) ? bk[cOut] : (cOut < 64 ? bq[cOut - 32] : bv[cOut - 64]);
        int kind = cOut >> 5, ch = cOut & 31;
        float* dst = (kind == 0) ? Kb : (kind == 1 ? Qb : Vb);
#pragma unroll
        for (int mi = 0; mi < 2; ++mi) {
#pragma unroll
            for (int r = 0; r < 8; ++r) {
                int p = tileM + mi * 16 + half * 8 + r;   // C/D layout: row = r + half*8
                int bb = p >> 11, n = p & 2047;
                dst[(ch * 2 + bb) * NSPAT + n] = acc[mi][ni][r] + bias;   // [c][b][n]
            }
        }
    }
}

// ---------------- WMMA GEMM 2: [4096 x 864] x [864 x 64] + bias + residual ----------------
__global__ void gemm2_out(const __bf16* __restrict__ A, const __bf16* __restrict__ Bt,
                          const float* __restrict__ ba, const float* __restrict__ x,
                          float* __restrict__ out) {
    int wave = threadIdx.x >> 5, lane = threadIdx.x & 31;
    int half = lane >> 4,        lcol = lane & 15;
    int tileM = (blockIdx.x * 8 + wave) * 32;
    int tileN = blockIdx.y * 32;

    const __bf16* pA0 = A  + (size_t)(tileM + lcol) * K2;
    const __bf16* pA1 = A  + (size_t)(tileM + 16 + lcol) * K2;
    const __bf16* pB0 = Bt + (size_t)(tileN + lcol) * K2;
    const __bf16* pB1 = Bt + (size_t)(tileN + 16 + lcol) * K2;

    v8f acc[2][2] = {{{}, {}}, {{}, {}}};
    for (int kb = 0; kb < K2; kb += 32) {
        v16bf a0 = load_afrag(pA0, kb, half);
        v16bf a1 = load_afrag(pA1, kb, half);
        v16bf b0 = load_bfrag(pB0, kb, half);
        v16bf b1 = load_bfrag(pB1, kb, half);
        __builtin_prefetch(pA0 + kb + 256, 0, 0);
        __builtin_prefetch(pA1 + kb + 256, 0, 0);
        acc[0][0] = __builtin_amdgcn_wmma_f32_16x16x32_bf16(false, a0, false, b0, (short)0, acc[0][0], false, false);
        acc[0][1] = __builtin_amdgcn_wmma_f32_16x16x32_bf16(false, a0, false, b1, (short)0, acc[0][1], false, false);
        acc[1][0] = __builtin_amdgcn_wmma_f32_16x16x32_bf16(false, a1, false, b0, (short)0, acc[1][0], false, false);
        acc[1][1] = __builtin_amdgcn_wmma_f32_16x16x32_bf16(false, a1, false, b1, (short)0, acc[1][1], false, false);
    }

#pragma unroll
    for (int ni = 0; ni < 2; ++ni) {
        int cOut = tileN + ni * 16 + lcol;     // 0..63
        float bias = ba[cOut];
#pragma unroll
        for (int mi = 0; mi < 2; ++mi) {
#pragma unroll
            for (int r = 0; r < 8; ++r) {
                int p = tileM + mi * 16 + half * 8 + r;
                int bb = p >> 11, n = p & 2047;
                size_t oi = (size_t)bb * (C2 * NSPAT) + (size_t)cOut * NSPAT + n;
                out[oi] = acc[mi][ni][r] + bias + x[oi];   // residual
            }
        }
    }
}

// ---------------- attention pass A: per-(c,m) column max & exp-sum; fold 1/Z into V ----------------
// S[c,n,m] = q[0,c,n]*k[0,c,m] + q[1,c,n]*k[1,c,m]; softmax over n (axis=1).
// Two-pass with 4-way split accumulators: 1 exp/element, no serial trans chains.
__global__ void attn_stats(const float* __restrict__ Q, const float* __restrict__ K,
                           float* __restrict__ V, float* __restrict__ Mcol) {
    int c = blockIdx.y;
    int m = blockIdx.x * 256 + threadIdx.x;
    __shared__ float q0s[NSPAT], q1s[NSPAT];
    const float* q0 = Q + (c * 2 + 0) * NSPAT;
    const float* q1 = Q + (c * 2 + 1) * NSPAT;
    for (int i = threadIdx.x; i < NSPAT; i += 256) { q0s[i] = q0[i]; q1s[i] = q1[i]; }
    __syncthreads();

    float k0 = K[(c * 2 + 0) * NSPAT + m];
    float k1 = K[(c * 2 + 1) * NSPAT + m];

    float mx[4] = {-INFINITY, -INFINITY, -INFINITY, -INFINITY};
    for (int n = 0; n < NSPAT; n += 4) {
#pragma unroll
        for (int j = 0; j < 4; ++j)
            mx[j] = fmaxf(mx[j], q0s[n + j] * k0 + q1s[n + j] * k1);
    }
    float M = fmaxf(fmaxf(mx[0], mx[1]), fmaxf(mx[2], mx[3]));

    float sm[4] = {0.f, 0.f, 0.f, 0.f};
    for (int n = 0; n < NSPAT; n += 4) {
#pragma unroll
        for (int j = 0; j < 4; ++j)
            sm[j] += __expf(q0s[n + j] * k0 + q1s[n + j] * k1 - M);
    }
    float sum = (sm[0] + sm[1]) + (sm[2] + sm[3]);

    Mcol[c * NSPAT + m] = M;
    float rz = 1.f / sum;
    V[(c * 2 + 0) * NSPAT + m] *= rz;   // fold normalization into v'
    V[(c * 2 + 1) * NSPAT + m] *= rz;
}

// ---------------- attention pass B: out[c,n,b] = sum_m exp(S - M[m]) * v'[m,b] ----------------
__global__ void attn_apply(const float* __restrict__ Q, const float* __restrict__ K,
                           const float* __restrict__ Vn, const float* __restrict__ Mcol,
                           float* __restrict__ Y) {
    int c = blockIdx.y;
    int n = blockIdx.x * 256 + threadIdx.x;
    __shared__ float k0s[NSPAT], k1s[NSPAT], v0s[NSPAT], v1s[NSPAT], ms[NSPAT]; // 40 KB
    for (int i = threadIdx.x; i < NSPAT; i += 256) {
        k0s[i] = K [(c * 2 + 0) * NSPAT + i];
        k1s[i] = K [(c * 2 + 1) * NSPAT + i];
        v0s[i] = Vn[(c * 2 + 0) * NSPAT + i];
        v1s[i] = Vn[(c * 2 + 1) * NSPAT + i];
        ms [i] = Mcol[c * NSPAT + i];
    }
    __syncthreads();

    float q0 = Q[(c * 2 + 0) * NSPAT + n];
    float q1 = Q[(c * 2 + 1) * NSPAT + n];
    float a0 = 0.f, a1 = 0.f, c0 = 0.f, c1 = 0.f;   // 2-way split accumulators
    for (int m = 0; m < NSPAT; m += 2) {
        float e0 = __expf(q0 * k0s[m]     + q1 * k1s[m]     - ms[m]);
        float e1 = __expf(q0 * k0s[m + 1] + q1 * k1s[m + 1] - ms[m + 1]);
        a0 += e0 * v0s[m];     a1 += e0 * v1s[m];
        c0 += e1 * v0s[m + 1]; c1 += e1 * v1s[m + 1];
    }
    // contiguous [C,N,B] buffer (== reinterpret as [B,C,D,H,W] for conv2)
    Y[(c * NSPAT + n) * 2 + 0] = a0 + c0;
    Y[(c * NSPAT + n) * 2 + 1] = a1 + c1;
}

// ---------------- launch ----------------
extern "C" void kernel_launch(void* const* d_in, const int* in_sizes, int n_in,
                              void* d_out, int out_size, void* d_ws, size_t ws_size,
                              hipStream_t stream) {
    (void)in_sizes; (void)n_in; (void)out_size; (void)ws_size;
    const float* x  = (const float*)d_in[0];
    const float* wk = (const float*)d_in[1];
    const float* bk = (const float*)d_in[2];
    const float* wq = (const float*)d_in[3];
    const float* bq = (const float*)d_in[4];
    const float* wv = (const float*)d_in[5];
    const float* bv = (const float*)d_in[6];
    const float* wa = (const float*)d_in[7];
    const float* ba = (const float*)d_in[8];
    float* out = (float*)d_out;

    char* w = (char*)d_ws;
    auto alloc = [&](size_t bytes) -> void* {
        void* p = (void*)w; w += (bytes + 255) & ~(size_t)255; return p;
    };
    __bf16* A1   = (__bf16*)alloc((size_t)MROWS * K1 * 2);   // 14.16 MB
    __bf16* A2   = (__bf16*)alloc((size_t)MROWS * K2 * 2);   //  7.08 MB
    __bf16* Wt1  = (__bf16*)alloc((size_t)N1 * K1 * 2);
    __bf16* Wt2  = (__bf16*)alloc((size_t)C2 * K2 * 2);
    float*  Qb   = (float*)alloc((size_t)C1 * 2 * NSPAT * 4);
    float*  Kb   = (float*)alloc((size_t)C1 * 2 * NSPAT * 4);
    float*  Vb   = (float*)alloc((size_t)C1 * 2 * NSPAT * 4);
    float*  Mcol = (float*)alloc((size_t)C1 * NSPAT * 4);
    float*  Yat  = (float*)alloc((size_t)C1 * NSPAT * 2 * 4);

    pack_w1<<<(N1 * K1 + 255) / 256, 256, 0, stream>>>(wk, wq, wv, Wt1);
    pack_w2<<<(C2 * K2 + 255) / 256, 256, 0, stream>>>(wa, Wt2);
    im2col1<<<((size_t)MROWS * K1 + 255) / 256, 256, 0, stream>>>(x, A1);

    gemm1_qkv<<<dim3(MROWS / 256, N1 / 32), 256, 0, stream>>>(A1, Wt1, bk, bq, bv, Kb, Qb, Vb);

    attn_stats<<<dim3(NSPAT / 256, C1), 256, 0, stream>>>(Qb, Kb, Vb, Mcol);
    attn_apply<<<dim3(NSPAT / 256, C1), 256, 0, stream>>>(Qb, Kb, Vb, Mcol, Yat);

    im2col2<<<((size_t)MROWS * K2 + 255) / 256, 256, 0, stream>>>(Yat, A2);

    gemm2_out<<<dim3(MROWS / 256, C2 / 32), 256, 0, stream>>>(A2, Wt2, ba, x, out);
}
